// SimpleGPT2_1683627180537
// MI455X (gfx1250) — compile-verified, matrix operand order checked by reading, not run on hardware
//
#include <hip/hip_runtime.h>
#include <stdint.h>

// ---------------- model constants (GPT-2 small, B=2, T=1024) ----------------
#define LAYERS 12
#define DMODEL 768
#define NHEAD  12
#define HDIM   64
#define VOCAB  50257
#define VPAD   50304            // vocab padded to multiple of 128 (and 8)
#define SEQ    1024
#define BATCHN 2
#define BT     (BATCHN*SEQ)     // 2048
#define D3     (3*DMODEL)       // 2304
#define D4     (4*DMODEL)       // 3072

typedef _Float16 half_t;
typedef unsigned long long u64;
typedef __attribute__((ext_vector_type(16))) _Float16 v16h;
typedef __attribute__((ext_vector_type(8)))  float    v8f;
typedef int v4i __attribute__((vector_size(16)));   // matches builtin param

union FragU { v16h v; uint32_t u[8]; };

// CDNA5 async global->LDS staging (ASYNCcnt-tracked), if the toolchain has it.
#if defined(__has_builtin)
#if __has_builtin(__builtin_amdgcn_global_load_async_to_lds_b128) && \
    __has_builtin(__builtin_amdgcn_s_wait_asynccnt)
#define USE_ASYNC_LDS 1
#endif
#endif

#ifdef USE_ASYNC_LDS
#define GAS __attribute__((address_space(1)))
#define LAS __attribute__((address_space(3)))
#endif

// ---------------------------------------------------------------------------
// Tiled WMMA GEMM: C[M,N](f32) = A[M,K](f16,row) x B[K,N](f16,row) + bias
// Requirements: M % 128 == 0, K % 32 == 0, lda % 8 == 0, ldb % 8 == 0.
// Block: 256 thr = 8 waves (2x4); block tile 128x128; wave tile 64x32
// (4x2 WMMA -> 8 v_wmma per wave per K-step). LDS double-buffered, one
// barrier per K-step. A tile staged via GLOBAL_LOAD_ASYNC_TO_LDS_B128
// (ASYNCcnt) when available; B tile register-transposed (needs reshape).
// ---------------------------------------------------------------------------
#define BM 128
#define BN 128
#define BK 32

__global__ __launch_bounds__(256) void wmma_gemm_kernel(
    const half_t* __restrict__ Ag, const half_t* __restrict__ Bg,
    const float* __restrict__ bias, float* __restrict__ Cg,
    int M, int N, int K, int lda, int ldb, int ldc,
    long long sA, long long sB, long long sC)
{
  __shared__ half_t As[2][BM][BK];   // 2 x 8 KB
  __shared__ half_t Bs[2][BN][BK];   // 2 x 8 KB, transposed: Bs[buf][n][k]

  const long long z = blockIdx.z;
  Ag += z * sA; Bg += z * sB; Cg += z * sC;

  const int m0 = blockIdx.y * BM;
  const int n0 = blockIdx.x * BN;
  const int tid  = threadIdx.x;
  const int lane = tid & 31;
  const int wid  = tid >> 5;
  const int wy   = wid >> 2;      // 0..1  (M direction, 64 rows each)
  const int wx   = wid & 3;       // 0..3  (N direction, 32 cols each)

  const int fr = lane & 15;       // fragment row (A) / col (B)
  const int kh = lane >> 4;       // K-half select (ISA 16-bit layout)

  // block-uniform: this block's N range is fully in bounds (scalar branch)
  const bool interior = (n0 + BN <= N);

  v8f acc[4][2];
  #pragma unroll
  for (int i = 0; i < 4; ++i)
    #pragma unroll
    for (int j = 0; j < 2; ++j)
      #pragma unroll
      for (int r = 0; r < 8; ++r) acc[i][j][r] = 0.0f;

  u64 bR[2][2];      // staged B micro-tiles: [slot][k-row 0/1]
  const uint16_t* B16 = reinterpret_cast<const uint16_t*>(Bg);

#ifdef USE_ASYNC_LDS
  // A tile: 512 x 16B async copies per tile (2 per thread), global -> LDS.
  auto stage_A = [&](int k0, int buf) {
    #pragma unroll
    for (int s = 0; s < 2; ++s) {
      int idx = tid + s * 256;                // 0..511
      int row = idx >> 2;                     // 0..127
      int col = (idx & 3) << 3;               // 0,8,16,24
      const half_t* gp = &Ag[(long long)(m0 + row) * lda + k0 + col];
      // generic->AS1: same numeric address; generic LDS low 32 bits = offset
      __builtin_amdgcn_global_load_async_to_lds_b128(
          (GAS v4i*)(uintptr_t)gp,
          (LAS v4i*)(uintptr_t)(uint32_t)(uintptr_t)&As[buf][row][col],
          0, 0);
    }
  };
#else
  uint4 aR[2];
  auto load_A = [&](int k0) {
    #pragma unroll
    for (int s = 0; s < 2; ++s) {
      int idx = tid + s * 256;
      int row = idx >> 2;
      int col = (idx & 3) << 3;
      aR[s] = *reinterpret_cast<const uint4*>(
          &Ag[(long long)(m0 + row) * lda + k0 + col]);
    }
  };
  auto store_A = [&](int buf) {
    #pragma unroll
    for (int s = 0; s < 2; ++s) {
      int idx = tid + s * 256;
      int row = idx >> 2;
      int col = (idx & 3) << 3;
      *reinterpret_cast<uint4*>(&As[buf][row][col]) = aR[s];
    }
  };
#endif

  auto load_B = [&](int k0) {
    if (interior) {
      #pragma unroll
      for (int s = 0; s < 2; ++s) {
        int idx = tid + s * 256;
        int gk  = k0 + ((idx >> 5) << 1);     // k pair
        int gn  = n0 + ((idx & 31) << 2);     // n quad
        bR[s][0] = *reinterpret_cast<const u64*>(&Bg[(long long)gk * ldb + gn]);
        bR[s][1] = *reinterpret_cast<const u64*>(&Bg[(long long)(gk + 1) * ldb + gn]);
      }
    } else {
      #pragma unroll
      for (int s = 0; s < 2; ++s) {
        int idx = tid + s * 256;
        int gk  = k0 + ((idx >> 5) << 1);
        int gn  = n0 + ((idx & 31) << 2);
        u64 lo = 0, hi = 0;
        #pragma unroll
        for (int j = 0; j < 4; ++j) {
          if (gn + j < N) {
            lo |= (u64)B16[(long long)gk * ldb + gn + j] << (16 * j);
            hi |= (u64)B16[(long long)(gk + 1) * ldb + gn + j] << (16 * j);
          }
        }
        bR[s][0] = lo; bR[s][1] = hi;
      }
    }
  };

  auto store_B = [&](int buf) {
    #pragma unroll
    for (int s = 0; s < 2; ++s) {
      int idx = tid + s * 256;
      int k = (idx >> 5) << 1;
      int n = (idx & 31) << 2;
      const uint16_t* lo = reinterpret_cast<const uint16_t*>(&bR[s][0]);
      const uint16_t* hi = reinterpret_cast<const uint16_t*>(&bR[s][1]);
      #pragma unroll
      for (int j = 0; j < 4; ++j) {
        uint32_t w = (uint32_t)lo[j] | ((uint32_t)hi[j] << 16);
        *reinterpret_cast<uint32_t*>(&Bs[buf][n + j][k]) = w;
      }
    }
  };

  // ---- prologue: stage tile 0 into buffer 0 ----
#ifdef USE_ASYNC_LDS
  stage_A(0, 0);
  load_B(0);
  store_B(0);
  __builtin_amdgcn_s_wait_asynccnt(0);
#else
  load_A(0);
  load_B(0);
  store_A(0);
  store_B(0);
#endif
  __syncthreads();

  int cur = 0;
  for (int k0 = 0; k0 < K; k0 += BK) {
    const bool has_next = (k0 + BK < K);
    if (has_next) {
#ifdef USE_ASYNC_LDS
      stage_A(k0 + BK, cur ^ 1);          // async copy into inactive buffer
      load_B(k0 + BK);
#else
      load_A(k0 + BK);
      load_B(k0 + BK);
#endif
      if (k0 + 2 * BK < K) {
        __builtin_prefetch(&Ag[(long long)(m0 + (tid >> 1)) * lda + k0 + 2 * BK], 0, 1);
        __builtin_prefetch(&Bg[(long long)(k0 + 2 * BK + (tid & 31)) * ldb + n0], 0, 1);
      }
    }

    // WMMAs on the current LDS buffer overlap in-flight copies.
    // Fragment layout: VGPR v holds K = (v/4)*16 + (lane/16)*8 + (v%4)*2 (+1)
    v16h a[4], b[2];
    #pragma unroll
    for (int i = 0; i < 4; ++i) {
      FragU f;
      int row = wy * 64 + i * 16 + fr;
      #pragma unroll
      for (int v = 0; v < 8; ++v) {
        int kk = ((v >> 2) << 4) + (kh << 3) + ((v & 3) << 1);
        f.u[v] = *reinterpret_cast<const uint32_t*>(&As[cur][row][kk]);
      }
      a[i] = f.v;
    }
    #pragma unroll
    for (int j = 0; j < 2; ++j) {
      FragU f;
      int col = wx * 32 + j * 16 + fr;
      #pragma unroll
      for (int v = 0; v < 8; ++v) {
        int kk = ((v >> 2) << 4) + (kh << 3) + ((v & 3) << 1);
        f.u[v] = *reinterpret_cast<const uint32_t*>(&Bs[cur][col][kk]);
      }
      b[j] = f.v;
    }
    #pragma unroll
    for (int i = 0; i < 4; ++i)
      #pragma unroll
      for (int j = 0; j < 2; ++j)
        acc[i][j] = __builtin_amdgcn_wmma_f32_16x16x32_f16(
            false, a[i], false, b[j], (short)0, acc[i][j], false, false);

    if (has_next) {
#ifdef USE_ASYNC_LDS
      store_B(cur ^ 1);
#else
      store_A(cur ^ 1);
      store_B(cur ^ 1);
#endif
    }
#ifdef USE_ASYNC_LDS
    __builtin_amdgcn_s_wait_asynccnt(0);  // A copies into cur^1 complete
#endif
    __syncthreads();
    cur ^= 1;
  }

  // store: lane n = lane%16, VGPR r -> M = r + 8*(lane/16)
  const int nl = lane & 15;
  const int hf = lane >> 4;
  #pragma unroll
  for (int i = 0; i < 4; ++i) {
    #pragma unroll
    for (int j = 0; j < 2; ++j) {
      int gn = n0 + wx * 32 + j * 16 + nl;
      if (gn >= N) continue;
      float bv = bias ? bias[gn] : 0.0f;
      #pragma unroll
      for (int r = 0; r < 8; ++r) {
        int gm = m0 + wy * 64 + i * 16 + hf * 8 + r;
        Cg[(long long)gm * ldc + gn] = acc[i][j][r] + bv;
      }
    }
  }
}

// ---------------------------------------------------------------------------
// Support kernels
// ---------------------------------------------------------------------------
__global__ void embed_kernel(const int* __restrict__ ids,
                             const float* __restrict__ wte,
                             const float* __restrict__ wpe,
                             float* __restrict__ x)
{
  int r = blockIdx.x;               // 0..BT-1
  int t = r % SEQ;
  int id = ids[r];
  for (int d = threadIdx.x; d < DMODEL; d += blockDim.x)
    x[(long long)r * DMODEL + d] =
        wte[(long long)id * DMODEL + d] + wpe[(long long)t * DMODEL + d];
}

__global__ __launch_bounds__(256) void layernorm_kernel(
    const float* __restrict__ x, const float* __restrict__ g,
    const float* __restrict__ b, half_t* __restrict__ out)
{
  __shared__ float red[256];
  __shared__ float s_mean, s_inv;
  const int r = blockIdx.x;
  const int tid = threadIdx.x;
  const float* row = x + (long long)r * DMODEL;

  float s = 0.f, s2 = 0.f;
  for (int d = tid; d < DMODEL; d += 256) { float v = row[d]; s += v; s2 += v * v; }
  red[tid] = s; __syncthreads();
  for (int o = 128; o > 0; o >>= 1) { if (tid < o) red[tid] += red[tid + o]; __syncthreads(); }
  if (tid == 0) s_mean = red[0] / DMODEL;
  __syncthreads();
  red[tid] = s2; __syncthreads();
  for (int o = 128; o > 0; o >>= 1) { if (tid < o) red[tid] += red[tid + o]; __syncthreads(); }
  if (tid == 0) {
    float var = red[0] / DMODEL - s_mean * s_mean;
    s_inv = rsqrtf(var + 1e-5f);
  }
  __syncthreads();
  float mean = s_mean, inv = s_inv;
  for (int d = tid; d < DMODEL; d += 256)
    out[(long long)r * DMODEL + d] = (half_t)((row[d] - mean) * inv * g[d] + b[d]);
}

__global__ void cast_f32_f16_kernel(const float* __restrict__ src,
                                    half_t* __restrict__ dst, long long n)
{
  long long i = (long long)blockIdx.x * blockDim.x + threadIdx.x;
  long long stride = (long long)gridDim.x * blockDim.x;
  for (; i < n; i += stride) dst[i] = (half_t)src[i];
}

// wteT[d][vp], vp in [0,VPAD); zero-fill the padded tail
__global__ void transpose_cast_kernel(const float* __restrict__ wte,
                                      half_t* __restrict__ wteT)
{
  long long i = (long long)blockIdx.x * blockDim.x + threadIdx.x;
  long long stride = (long long)gridDim.x * blockDim.x;
  const long long n = (long long)DMODEL * VPAD;
  for (; i < n; i += stride) {
    long long d = i / VPAD, vp = i % VPAD;
    wteT[i] = (vp < VOCAB) ? (half_t)wte[vp * DMODEL + d] : (half_t)0.0f;
  }
}

__global__ void split_qkv_kernel(const float* __restrict__ qkv,
                                 half_t* __restrict__ q,   // [B,H,T,HD]
                                 half_t* __restrict__ kT,  // [B,H,HD,T]
                                 half_t* __restrict__ v)   // [B,H,T,HD]
{
  long long i = (long long)blockIdx.x * blockDim.x + threadIdx.x;
  long long stride = (long long)gridDim.x * blockDim.x;
  const long long n = (long long)BT * D3;
  for (; i < n; i += stride) {
    int r = (int)(i / D3), c = (int)(i % D3);
    int b = r / SEQ, t = r % SEQ;
    int which = c / DMODEL;
    int h = (c % DMODEL) / HDIM;
    int d = c % HDIM;
    half_t val = (half_t)qkv[i];
    long long bh = (long long)(b * NHEAD + h);
    if (which == 0)      q [(bh * SEQ + t) * HDIM + d] = val;
    else if (which == 2) v [(bh * SEQ + t) * HDIM + d] = val;
    else                 kT[(bh * HDIM + d) * SEQ + t] = val;
  }
}

__global__ __launch_bounds__(256) void softmax_causal_kernel(
    const float* __restrict__ scores, half_t* __restrict__ att, float scale)
{
  __shared__ float red[256];
  const long long rb = blockIdx.x;          // 0 .. B*H*T-1
  const int i = (int)(rb % SEQ);            // query position
  const int tid = threadIdx.x;
  const float* srow = scores + rb * SEQ;
  half_t* arow = att + rb * SEQ;

  float m = -3.0e38f;
  for (int j = tid; j <= i; j += 256) m = fmaxf(m, srow[j] * scale);
  red[tid] = m; __syncthreads();
  for (int o = 128; o > 0; o >>= 1) { if (tid < o) red[tid] = fmaxf(red[tid], red[tid + o]); __syncthreads(); }
  float mx = red[0]; __syncthreads();

  float s = 0.f;
  for (int j = tid; j <= i; j += 256) s += expf(srow[j] * scale - mx);
  red[tid] = s; __syncthreads();
  for (int o = 128; o > 0; o >>= 1) { if (tid < o) red[tid] += red[tid + o]; __syncthreads(); }
  float inv = 1.0f / red[0];

  for (int j = tid; j < SEQ; j += 256) {
    float o = (j <= i) ? expf(srow[j] * scale - mx) * inv : 0.0f;
    arow[j] = (half_t)o;
  }
}

__global__ void merge_heads_kernel(const float* __restrict__ yh, // [B,H,T,HD]
                                   half_t* __restrict__ out)     // [BT, D]
{
  long long i = (long long)blockIdx.x * blockDim.x + threadIdx.x;
  long long stride = (long long)gridDim.x * blockDim.x;
  const long long n = (long long)BT * DMODEL;
  for (; i < n; i += stride) {
    int r = (int)(i / DMODEL), c = (int)(i % DMODEL);
    int b = r / SEQ, t = r % SEQ;
    int h = c / HDIM, d = c % HDIM;
    out[i] = (half_t)yh[(((long long)(b * NHEAD + h)) * SEQ + t) * HDIM + d];
  }
}

__global__ void gelu_kernel(const float* __restrict__ in,
                            half_t* __restrict__ out, long long n)
{
  long long i = (long long)blockIdx.x * blockDim.x + threadIdx.x;
  long long stride = (long long)gridDim.x * blockDim.x;
  for (; i < n; i += stride) {
    float v = in[i];
    out[i] = (half_t)(0.5f * v * (1.0f + erff(v * 0.70710678118654752f)));
  }
}

__global__ void add_inplace_kernel(float* __restrict__ x,
                                   const float* __restrict__ y, long long n)
{
  long long i = (long long)blockIdx.x * blockDim.x + threadIdx.x;
  long long stride = (long long)gridDim.x * blockDim.x;
  for (; i < n; i += stride) x[i] += y[i];
}

// ---------------------------------------------------------------------------
// Host-side orchestration
// ---------------------------------------------------------------------------
static inline void launch_gemm(hipStream_t s,
                               const half_t* A, const half_t* B,
                               const float* bias, float* C,
                               int M, int N, int K, int lda, int ldb, int ldc,
                               long long sA = 0, long long sB = 0,
                               long long sC = 0, int batch = 1)
{
  dim3 grid((N + BN - 1) / BN, M / BM, batch);
  wmma_gemm_kernel<<<grid, dim3(256), 0, s>>>(A, B, bias, C, M, N, K,
                                              lda, ldb, ldc, sA, sB, sC);
}

extern "C" void kernel_launch(void* const* d_in, const int* in_sizes, int n_in,
                              void* d_out, int out_size, void* d_ws, size_t ws_size,
                              hipStream_t stream)
{
  (void)in_sizes; (void)n_in; (void)out_size; (void)ws_size;
  const int*   ids    = (const int*)  d_in[0];
  const float* wte    = (const float*)d_in[1];
  const float* wpe    = (const float*)d_in[2];
  const float* ln1_g  = (const float*)d_in[3];
  const float* ln1_b  = (const float*)d_in[4];
  const float* attn_w = (const float*)d_in[5];
  const float* attn_b = (const float*)d_in[6];
  const float* proj_w = (const float*)d_in[7];
  const float* proj_b = (const float*)d_in[8];
  const float* ln2_g  = (const float*)d_in[9];
  const float* ln2_b  = (const float*)d_in[10];
  const float* fc_w   = (const float*)d_in[11];
  const float* fc_b   = (const float*)d_in[12];
  const float* fc2_w  = (const float*)d_in[13];
  const float* fc2_b  = (const float*)d_in[14];
  const float* lnf_g  = (const float*)d_in[15];
  const float* lnf_b  = (const float*)d_in[16];
  float* out = (float*)d_out;

  // ---- bump allocator over workspace ----
  char* cur = (char*)d_ws;
  auto alloc = [&](size_t bytes) -> void* {
    void* r = (void*)cur;
    cur += (bytes + 255) & ~(size_t)255;
    return r;
  };
  half_t* wteT   = (half_t*)alloc((size_t)DMODEL * VPAD * 2); // [D, VPAD]
  float*  x      = (float*) alloc((size_t)BT * DMODEL * 4);
  half_t* h      = (half_t*)alloc((size_t)BT * DMODEL * 2);
  float*  qkv    = (float*) alloc((size_t)BT * D3 * 4);
  half_t* qb     = (half_t*)alloc((size_t)BT * DMODEL * 2);
  half_t* kTb    = (half_t*)alloc((size_t)BT * DMODEL * 2);
  half_t* vb     = (half_t*)alloc((size_t)BT * DMODEL * 2);
  float*  scores = (float*) alloc((size_t)BATCHN * NHEAD * SEQ * SEQ * 4);
  half_t* att    = (half_t*)alloc((size_t)BATCHN * NHEAD * SEQ * SEQ * 2);
  float*  yheads = (float*) alloc((size_t)BT * DMODEL * 4);
  half_t* ybf    = (half_t*)alloc((size_t)BT * DMODEL * 2);
  float*  tmp    = (float*) alloc((size_t)BT * D4 * 4);
  half_t* gelub  = (half_t*)alloc((size_t)BT * D4 * 2);
  half_t* wA     = (half_t*)alloc((size_t)DMODEL * D3 * 2);
  half_t* wP     = (half_t*)alloc((size_t)DMODEL * DMODEL * 2);
  half_t* wF     = (half_t*)alloc((size_t)DMODEL * D4 * 2);
  half_t* wF2    = (half_t*)alloc((size_t)D4 * DMODEL * 2);

  const float attn_scale = 0.125f;              // 1/sqrt(64)
  const int EB = 4096;                          // elementwise grid blocks

  // tied head weight, transposed + cast (recomputed each call: deterministic)
  transpose_cast_kernel<<<EB, 256, 0, stream>>>(wte, wteT);

  // embeddings
  embed_kernel<<<BT, 256, 0, stream>>>(ids, wte, wpe, x);

  for (int l = 0; l < LAYERS; ++l) {
    // ---- attention ----
    layernorm_kernel<<<BT, 256, 0, stream>>>(x, ln1_g + l * DMODEL,
                                             ln1_b + l * DMODEL, h);
    cast_f32_f16_kernel<<<EB, 256, 0, stream>>>(attn_w + (size_t)l * DMODEL * D3,
                                                wA, (long long)DMODEL * D3);
    launch_gemm(stream, h, wA, attn_b + l * D3, qkv,
                BT, D3, DMODEL, DMODEL, D3, D3);
    split_qkv_kernel<<<EB, 256, 0, stream>>>(qkv, qb, kTb, vb);

    // scores[b,h] = q[T,HD] x kT[HD,T], batched over B*H
    launch_gemm(stream, qb, kTb, nullptr, scores,
                SEQ, SEQ, HDIM, HDIM, SEQ, SEQ,
                (long long)SEQ * HDIM, (long long)HDIM * SEQ,
                (long long)SEQ * SEQ, BATCHN * NHEAD);
    softmax_causal_kernel<<<BATCHN * NHEAD * SEQ, 256, 0, stream>>>(
        scores, att, attn_scale);

    // yheads[b,h] = att[T,T] x v[T,HD]
    launch_gemm(stream, att, vb, nullptr, yheads,
                SEQ, HDIM, SEQ, SEQ, HDIM, HDIM,
                (long long)SEQ * SEQ, (long long)SEQ * HDIM,
                (long long)SEQ * HDIM, BATCHN * NHEAD);
    merge_heads_kernel<<<EB, 256, 0, stream>>>(yheads, ybf);

    cast_f32_f16_kernel<<<EB, 256, 0, stream>>>(proj_w + (size_t)l * DMODEL * DMODEL,
                                                wP, (long long)DMODEL * DMODEL);
    launch_gemm(stream, ybf, wP, proj_b + l * DMODEL, tmp,
                BT, DMODEL, DMODEL, DMODEL, DMODEL, DMODEL);
    add_inplace_kernel<<<EB, 256, 0, stream>>>(x, tmp, (long long)BT * DMODEL);

    // ---- MLP ----
    layernorm_kernel<<<BT, 256, 0, stream>>>(x, ln2_g + l * DMODEL,
                                             ln2_b + l * DMODEL, h);
    cast_f32_f16_kernel<<<EB, 256, 0, stream>>>(fc_w + (size_t)l * DMODEL * D4,
                                                wF, (long long)DMODEL * D4);
    launch_gemm(stream, h, wF, fc_b + l * D4, tmp,
                BT, D4, DMODEL, DMODEL, D4, D4);
    gelu_kernel<<<EB, 256, 0, stream>>>(tmp, gelub, (long long)BT * D4);
    cast_f32_f16_kernel<<<EB, 256, 0, stream>>>(fc2_w + (size_t)l * D4 * DMODEL,
                                                wF2, (long long)D4 * DMODEL);
    launch_gemm(stream, gelub, wF2, fc2_b + l * DMODEL, tmp,
                BT, DMODEL, D4, D4, DMODEL, DMODEL);
    add_inplace_kernel<<<EB, 256, 0, stream>>>(x, tmp, (long long)BT * DMODEL);
  }

  // final LN + tied-head logits (B has padded stride VPAD, store clamps to VOCAB)
  layernorm_kernel<<<BT, 256, 0, stream>>>(x, lnf_g, lnf_b, h);
  launch_gemm(stream, h, wteT, nullptr, out,
              BT, VOCAB, DMODEL, DMODEL, VPAD, VOCAB);
}
